// TransformerDecoderLayerWithCache_8650064134646
// MI455X (gfx1250) — matrix operand order repaired and checked by reading, not compiled, not run on hardware
//
#include <hip/hip_runtime.h>
#include <hip/hip_bf16.h>

typedef __attribute__((ext_vector_type(16))) _Float16 v16h;
typedef __attribute__((ext_vector_type(8)))  float    v8f;
typedef __attribute__((ext_vector_type(4)))  float    floatx4;

#define NB    8      // batch
#define NH    8      // heads
#define DH    256    // per-head dim (== d_model)
#define BH    (NB*NH)
#define SPAST 4096
#define S1    (SPAST+1)
#define DFF   1024
#define DTOT  (NH*DH)
#define VSPLIT 16    // S-dim splits for the attn*V pass

// ---------------------------------------------------------------------------
// Pad an (8 x K) f32 matrix to (16 x K) f16, rows 8..15 zero.
// ---------------------------------------------------------------------------
__global__ void pad16_f16(const float* __restrict__ in, _Float16* __restrict__ out, int K) {
    int idx = blockIdx.x * 256 + threadIdx.x;
    if (idx < 16 * K) {
        int r = idx / K, c = idx - r * K;
        out[idx] = (r < NB) ? (_Float16)in[r * K + c] : (_Float16)0.0f;
    }
}

// ---------------------------------------------------------------------------
// WMMA GEMM: out(8,N) = A16(16,K,f16; rows 8..15 zero) @ W(K,N,f32) + bias
// One wave per 16-wide N tile, looping K in steps of 32.
// Uses probe-confirmed __builtin_amdgcn_wmma_f32_16x16x32_f16.
// ---------------------------------------------------------------------------
__global__ void gemm_wmma_f16(const _Float16* __restrict__ A16,
                              const float* __restrict__ W,
                              const float* __restrict__ bias,
                              float* __restrict__ out,
                              int N, int K, int relu) {
    int wave = threadIdx.x >> 5;
    int lane = threadIdx.x & 31;
    int n0 = (blockIdx.x * (blockDim.x >> 5) + wave) * 16;
    if (n0 >= N) return;                       // wave-uniform: EXEC stays full
    int half = lane >> 4;                      // 0: lanes 0-15, 1: lanes 16-31
    int m    = lane & 15;                      // A row (for lanes in its half)
    int n    = n0 + m;                         // B column
    v8f c = {};
    for (int k0 = 0; k0 < K; k0 += 32) {
        v16h a, b;
        // A fragment: ISA 7.12.2 16-bit A 16x32 layout
        #pragma unroll
        for (int i = 0; i < 16; ++i) {
            int v  = i >> 1;
            int kl = ((v & 4) ? 16 : 0) + (v & 3) * 2 + half * 8 + (i & 1);
            a[i] = A16[m * K + k0 + kl];
        }
        // B fragment: same K order as A (lanes 0-15: K=0..7,16..23; lanes 16-31: K=8..15,24..31)
        #pragma unroll
        for (int i = 0; i < 16; ++i) {
            b[i] = (_Float16)W[(size_t)(k0 + (i >> 3) * 16 + half * 8 + (i & 7)) * N + n];
        }
        c = __builtin_amdgcn_wmma_f32_16x16x32_f16(false, a, false, b,
                                                   (short)0, c, false, false);
    }
    // C layout: lane (0-15) = N, VGPR r = M row r (lanes 0-15) / r+8 (16-31).
    // Batch rows 0..7 live entirely in lanes 0..15.
    if (half == 0) {
        float bv = bias[n];
        #pragma unroll
        for (int r = 0; r < NB; ++r) {
            float v = c[r] + bv;
            if (relu) v = fmaxf(v, 0.0f);
            out[(size_t)r * N + n] = v;
        }
    }
}

// ---------------------------------------------------------------------------
// Attention scores + fused K-cache write-out.
// grid (chunks, BH). One wave per key row: 32 lanes x 8 floats (2x b128 loads),
// intra-wave shuffle reduce. Optionally copies each K row to the output cache.
// ---------------------------------------------------------------------------
__global__ void attn_scores(const float* __restrict__ qbase, int qbstride,
                            const float* __restrict__ kpast,
                            const float* __restrict__ knewbase, int nbstride,
                            float* __restrict__ scores, int sstride,
                            float* __restrict__ kout,
                            int S_past, int S_total, float scale) {
    int bh = blockIdx.y;
    int b = bh >> 3, h = bh & 7;
    __shared__ floatx4 qs[DH / 4];
    int t = threadIdx.x;
    if (t < DH / 4)
        qs[t] = ((const floatx4*)(qbase + (size_t)b * qbstride + h * DH))[t];
    __syncthreads();
    int lane = t & 31, wave = t >> 5;
    int wglobal = blockIdx.x * (blockDim.x >> 5) + wave;
    int nwaves  = gridDim.x * (blockDim.x >> 5);
    for (int s = wglobal; s < S_total; s += nwaves) {
        const float* krow = (s < S_past)
            ? (kpast + ((size_t)bh * S_past + s) * DH)
            : (knewbase + (size_t)b * nbstride + h * DH);
        floatx4 k0 = ((const floatx4*)krow)[lane * 2];
        floatx4 k1 = ((const floatx4*)krow)[lane * 2 + 1];
        if (kout) {
            floatx4* o = (floatx4*)(kout + ((size_t)bh * S_total + s) * DH);
            o[lane * 2] = k0; o[lane * 2 + 1] = k1;
        }
        floatx4 q0 = qs[lane * 2], q1 = qs[lane * 2 + 1];
        float p = k0.x * q0.x + k0.y * q0.y + k0.z * q0.z + k0.w * q0.w
                + k1.x * q1.x + k1.y * q1.y + k1.z * q1.z + k1.w * q1.w;
        #pragma unroll
        for (int off = 16; off > 0; off >>= 1)
            p += __shfl_xor(p, off, 32);
        if (lane == 0) scores[(size_t)bh * sstride + s] = p * scale;
    }
}

// ---------------------------------------------------------------------------
// Softmax over S scores per (b,h). One block per bh.
// ---------------------------------------------------------------------------
__global__ void softmax_k(float* __restrict__ scores, int sstride, int S) {
    int bh = blockIdx.x;
    float* sc = scores + (size_t)bh * sstride;
    __shared__ float red[256];
    int t = threadIdx.x;
    float mx = -3.0e38f;
    for (int s = t; s < S; s += 256) mx = fmaxf(mx, sc[s]);
    red[t] = mx; __syncthreads();
    for (int o = 128; o > 0; o >>= 1) { if (t < o) red[t] = fmaxf(red[t], red[t + o]); __syncthreads(); }
    mx = red[0]; __syncthreads();
    float sum = 0.0f;
    for (int s = t; s < S; s += 256) { float e = __expf(sc[s] - mx); sc[s] = e; sum += e; }
    red[t] = sum; __syncthreads();
    for (int o = 128; o > 0; o >>= 1) { if (t < o) red[t] += red[t + o]; __syncthreads(); }
    float inv = 1.0f / red[0]; __syncthreads();
    for (int s = t; s < S; s += 256) sc[s] *= inv;
}

// ---------------------------------------------------------------------------
// Partial probs.V over an S-slice, fused with V-cache write-out.
// grid (VSPLIT, BH). Thread t: row-group g=t/64 (4 rows in flight), columns
// (t%64)*4 .. +3 via b128 loads; cross-group LDS reduce; partial to scratch.
// ---------------------------------------------------------------------------
__global__ void attn_av_part(const float* __restrict__ probs, int sstride,
                             const float* __restrict__ vpast,
                             float* __restrict__ partial,   // (VSPLIT, BH, DH)
                             float* __restrict__ vout,
                             int S_past, int S_total) {
    int bh = blockIdx.y;
    int split = blockIdx.x;
    int sper = S_past / gridDim.x;
    int s0 = split * sper;
    int t = threadIdx.x;
    int g = t >> 6;              // 0..3
    int c4 = t & 63;             // column/4
    const float* pv = vpast + (size_t)bh * S_past * DH;
    const float* pr = probs + (size_t)bh * sstride;
    __shared__ float pl[128];
    floatx4 acc = {0.0f, 0.0f, 0.0f, 0.0f};
    for (int base = s0; base < s0 + sper; base += 128) {
        int cnt = (s0 + sper - base < 128) ? (s0 + sper - base) : 128;
        if (t < cnt) pl[t] = pr[base + t];
        if (t == 0) __builtin_prefetch(pr + base + 128, 0, 0);  // next prob chunk
        __syncthreads();
        #pragma unroll 4
        for (int i = g; i < cnt; i += 4) {
            floatx4 vv = ((const floatx4*)(pv + (size_t)(base + i) * DH))[c4];
            if (vout)
                ((floatx4*)(vout + ((size_t)bh * S_total + base + i) * DH))[c4] = vv;
            acc += pl[i] * vv;
        }
        __syncthreads();
    }
    __shared__ floatx4 red4[256];
    red4[t] = acc; __syncthreads();
    if (g == 0) {
        acc = red4[t] + red4[t + 64] + red4[t + 128] + red4[t + 192];
        ((floatx4*)(partial + ((size_t)split * BH + bh) * DH))[c4] = acc;
    }
}

// ---------------------------------------------------------------------------
// Sum the VSPLIT partials, fold in the appended v_new row (and write it to the
// output cache), produce y(bh,256). One block per bh.
// ---------------------------------------------------------------------------
__global__ void attn_av_reduce(const float* __restrict__ partial, int nsplit,
                               const float* __restrict__ probs, int sstride,
                               const float* __restrict__ vnewbase, int nbstride,
                               float* __restrict__ y, float* __restrict__ vout,
                               int S_past, int S_total) {
    int bh = blockIdx.x;
    int t = threadIdx.x;
    float acc = 0.0f;
    #pragma unroll 4
    for (int sp = 0; sp < nsplit; ++sp)
        acc += partial[((size_t)sp * BH + bh) * DH + t];
    if (S_total > S_past) {
        int b = bh >> 3, h = bh & 7;
        float vv = vnewbase[(size_t)b * nbstride + h * DH + t];
        if (vout) vout[((size_t)bh * S_total + S_past) * DH + t] = vv;
        acc += probs[(size_t)bh * sstride + S_past] * vv;
    }
    y[(size_t)bh * DH + t] = acc;
}

// ---------------------------------------------------------------------------
// LayerNorm(pre + res) over last dim of 256, no affine. One block per row.
// ---------------------------------------------------------------------------
__global__ void ln_res(const float* __restrict__ pre, const float* __restrict__ res,
                       float* __restrict__ out) {
    int b = blockIdx.x;
    int t = threadIdx.x;
    float v = pre[b * DH + t] + res[b * DH + t];
    __shared__ float red[256];
    red[t] = v; __syncthreads();
    for (int o = 128; o > 0; o >>= 1) { if (t < o) red[t] += red[t + o]; __syncthreads(); }
    float mean = red[0] * (1.0f / DH); __syncthreads();
    float d = v - mean;
    red[t] = d * d; __syncthreads();
    for (int o = 128; o > 0; o >>= 1) { if (t < o) red[t] += red[t + o]; __syncthreads(); }
    float var = red[0] * (1.0f / DH);
    out[b * DH + t] = d * __frsqrt_rn(var + 1e-5f);
}

extern "C" void kernel_launch(void* const* d_in, const int* in_sizes, int n_in,
                              void* d_out, int out_size, void* d_ws, size_t ws_size,
                              hipStream_t stream) {
    const float* x      = (const float*)d_in[0];
    const float* past_k = (const float*)d_in[1];
    const float* past_v = (const float*)d_in[2];
    const float* enc_k  = (const float*)d_in[3];
    const float* enc_v  = (const float*)d_in[4];
    const float* Wqkv   = (const float*)d_in[5];
    const float* bqkv   = (const float*)d_in[6];
    const float* Wo     = (const float*)d_in[7];
    const float* bo     = (const float*)d_in[8];
    const float* Wqc    = (const float*)d_in[9];
    const float* bqc    = (const float*)d_in[10];
    const float* Woc    = (const float*)d_in[11];
    const float* boc    = (const float*)d_in[12];
    const float* W1     = (const float*)d_in[13];
    const float* b1     = (const float*)d_in[14];
    const float* W2     = (const float*)d_in[15];
    const float* b2     = (const float*)d_in[16];

    float* outv  = (float*)d_out;                       // (8,1,256) = 2048
    float* k_out = outv + (size_t)NB * DH;              // (8,8,4097,256)
    float* v_out = k_out + (size_t)BH * S1 * DH;        // (8,8,4097,256)

    char* ws = (char*)d_ws;
    _Float16* a16   = (_Float16*)(ws + 0);              //  64 KB (16 x 2048 f16)
    float* qkv      = (float*)(ws +   65536);           // 192 KB (8 x 6144)
    float* yatt     = (float*)(ws +  262144);           //  64 KB (8 x 2048)
    float* y1       = (float*)(ws +  327680);           //   8 KB (8 x 256)
    float* qc       = (float*)(ws +  335872);           //  64 KB (8 x 2048)
    float* yc       = (float*)(ws +  401408);           //  64 KB (8 x 2048)
    float* y2       = (float*)(ws +  466944);           //   8 KB (8 x 256)
    float* hbuf     = (float*)(ws +  475136);           //  32 KB (8 x 1024)
    float* pre      = (float*)(ws +  507904);           //   8 KB (8 x 256)
    float* scores   = (float*)(ws +  516096);           //  ~1 MB (64 x 4104)
    float* partials = (float*)(ws + 1566720);           //   1 MB (16 x 64 x 256)
    const int SSTRIDE = 4104;
    const float scale = 1.0f / 16.0f;                   // 1/sqrt(256)

    dim3 blk256(256), blk128(128);

    // ---- qkv projection: (8,256) @ (256,6144) ----
    pad16_f16<<<dim3((16 * DH + 255) / 256), blk256, 0, stream>>>(x, a16, DH);
    gemm_wmma_f16<<<dim3(3 * DTOT / 64), blk128, 0, stream>>>(a16, Wqkv, bqkv, qkv, 3 * DTOT, DH, 0);

    // ---- self-attention over S1=4097 keys; fused K/V cache write-out ----
    attn_scores<<<dim3(64, BH), blk256, 0, stream>>>(
        qkv, 3 * DTOT, past_k, qkv + DTOT, 3 * DTOT,
        scores, SSTRIDE, k_out, SPAST, S1, scale);
    softmax_k<<<dim3(BH), blk256, 0, stream>>>(scores, SSTRIDE, S1);
    attn_av_part<<<dim3(VSPLIT, BH), blk256, 0, stream>>>(
        scores, SSTRIDE, past_v, partials, v_out, SPAST, S1);
    attn_av_reduce<<<dim3(BH), blk256, 0, stream>>>(
        partials, VSPLIT, scores, SSTRIDE, qkv + 2 * DTOT, 3 * DTOT,
        yatt, v_out, SPAST, S1);

    // ---- out-proj + residual + LN ----
    pad16_f16<<<dim3((16 * DTOT + 255) / 256), blk256, 0, stream>>>(yatt, a16, DTOT);
    gemm_wmma_f16<<<dim3(DH / 64), blk128, 0, stream>>>(a16, Wo, bo, pre, DH, DTOT, 0);
    ln_res<<<dim3(NB), blk256, 0, stream>>>(pre, x, y1);

    // ---- cross-attention ----
    pad16_f16<<<dim3((16 * DH + 255) / 256), blk256, 0, stream>>>(y1, a16, DH);
    gemm_wmma_f16<<<dim3(DTOT / 64), blk128, 0, stream>>>(a16, Wqc, bqc, qc, DTOT, DH, 0);
    attn_scores<<<dim3(64, BH), blk256, 0, stream>>>(
        qc, DTOT, enc_k, nullptr, 0,
        scores, SSTRIDE, nullptr, SPAST, SPAST, scale);
    softmax_k<<<dim3(BH), blk256, 0, stream>>>(scores, SSTRIDE, SPAST);
    attn_av_part<<<dim3(VSPLIT, BH), blk256, 0, stream>>>(
        scores, SSTRIDE, enc_v, partials, nullptr, SPAST, SPAST);
    attn_av_reduce<<<dim3(BH), blk256, 0, stream>>>(
        partials, VSPLIT, scores, SSTRIDE, nullptr, 0,
        yc, nullptr, SPAST, SPAST);
    pad16_f16<<<dim3((16 * DTOT + 255) / 256), blk256, 0, stream>>>(yc, a16, DTOT);
    gemm_wmma_f16<<<dim3(DH / 64), blk128, 0, stream>>>(a16, Woc, boc, pre, DH, DTOT, 0);
    ln_res<<<dim3(NB), blk256, 0, stream>>>(pre, y1, y2);

    // ---- FFN ----
    pad16_f16<<<dim3((16 * DH + 255) / 256), blk256, 0, stream>>>(y2, a16, DH);
    gemm_wmma_f16<<<dim3(DFF / 64), blk128, 0, stream>>>(a16, W1, b1, hbuf, DFF, DH, 1);
    pad16_f16<<<dim3((16 * DFF + 255) / 256), blk256, 0, stream>>>(hbuf, a16, DFF);
    gemm_wmma_f16<<<dim3(DH / 64), blk128, 0, stream>>>(a16, W2, b2, pre, DH, DFF, 0);
    ln_res<<<dim3(NB), blk256, 0, stream>>>(pre, y2, outv);
}
